// GraphTransformer_63969242906718
// MI455X (gfx1250) — matrix operand order repaired and checked by reading, not compiled, not run on hardware
//
#include <hip/hip_runtime.h>

typedef __attribute__((ext_vector_type(16))) _Float16 v16h;
typedef __attribute__((ext_vector_type(8)))  float    v8f;

// ---- problem constants (match reference) ----
#define NL    4
#define DIM   512
#define HGC   8
#define CGC   64
#define HTC   4
#define CTC   128
#define FFD   2048
#define NG    128
#define NPGC  256
#define NN    (NG * NPGC)      // 32768 nodes
#define NE    (NN * 16)        // 524288 edges
#define AFN   9
#define AVN   64
#define BFN   3
#define BVN   8
#define OUTD  128

// =====================================================================
// WMMA GEMM: C[M,N] = act( scale * A[M,K] @ B^T[N,K] + bias[N] )
// B is ALWAYS stored transposed [N][K] (weights pre-transposed once; K/vT
// naturally in this layout), so A and B staging are both contiguous 32B
// chunks -> coalesced global_load_b128, ds_store_b128, ds_load_b128.
// Block tile 128x128, BK=32; 8 wave32 waves, each 2x4 WMMA accumulators.
// K loop software-pipelined (next tile fetched to regs during WMMAs).
// AF32: A is f32 (converted at stage). OUTF32: C stored f32 else f16.
// Batched via blockIdx.z: off = (z/inner)*s1 + (z%inner)*s2.
// =====================================================================
template <bool AF32, bool OUTF32, bool RELU>
__global__ __launch_bounds__(256) void hgemm(
    const void* __restrict__ Ap, const _Float16* __restrict__ B,
    const float* __restrict__ bias, void* __restrict__ Cp, int K, long lda,
    long ldb, long ldc, int inner, long sA1, long sA2, long sB1, long sB2,
    long sC1, long sC2, float scale) {
  constexpr int ASTR = 40;   // halfs per A-tile row (32 k + 8 pad)
  constexpr int BSTRT = 40;  // halfs per B-tile column (32 k + 8 pad)
  __shared__ _Float16 As[128 * ASTR];   // [row][k]
  __shared__ _Float16 Bt[128 * BSTRT];  // [col][k]

  const int t = threadIdx.x;
  const int lane = t & 31;
  const int wid = t >> 5;
  const int wm = wid & 3;   // 0..3 -> 32-row slice
  const int wn = wid >> 2;  // 0..1 -> 64-col slice
  const int lr = lane & 15;
  const bool hi = lane >= 16;

  const int bz = blockIdx.z;
  const long offA = (long)(bz / inner) * sA1 + (long)(bz % inner) * sA2;
  const long offB = (long)(bz / inner) * sB1 + (long)(bz % inner) * sB2;
  const long offC = (long)(bz / inner) * sC1 + (long)(bz % inner) * sC2;

  const long rowBase = (long)blockIdx.y * 128;
  const long colBase = (long)blockIdx.x * 128;

  // staging ownership
  const int ar = t >> 1, ac = (t & 1) * 16;    // A: row 0..127, 16-k chunk
  const int bn = t & 127, bk = (t >> 7) * 16;  // B: col 0..127, 16-k chunk

  v8f acc[2][4] = {};

  _Float16 abuf[16], bbuf[16];
  auto fetchA = [&](int kbase) {
    if (AF32) {
      const float* sa = (const float*)Ap + offA + (rowBase + ar) * lda + kbase + ac;
#pragma unroll
      for (int j = 0; j < 16; ++j) abuf[j] = (_Float16)sa[j];
    } else {
      const _Float16* sa =
          (const _Float16*)Ap + offA + (rowBase + ar) * lda + kbase + ac;
#pragma unroll
      for (int j = 0; j < 16; ++j) abuf[j] = sa[j];
    }
  };
  auto fetchB = [&](int kbase) {
    const _Float16* sb = B + offB + (colBase + bn) * ldb + kbase + bk;
#pragma unroll
    for (int j = 0; j < 16; ++j) bbuf[j] = sb[j];
  };

  fetchA(0);
  fetchB(0);

  for (int kk = 0; kk < K; kk += 32) {
    {  // commit staged tile to LDS (contiguous -> ds_store_b128)
      _Float16* da = &As[ar * ASTR + ac];
#pragma unroll
      for (int j = 0; j < 16; ++j) da[j] = abuf[j];
      _Float16* db = &Bt[bn * BSTRT + bk];
#pragma unroll
      for (int j = 0; j < 16; ++j) db[j] = bbuf[j];
    }
    __syncthreads();

    // pipeline: fetch next K-tile from global while WMMAs run
    if (kk + 32 < K) {
      fetchA(kk + 32);
      fetchB(kk + 32);
    }

    // fragments per ISA 7.12.2 layout tables (all contiguous 16B runs)
    v16h af[2], bf[4];
#pragma unroll
    for (int mt = 0; mt < 2; ++mt) {
      const _Float16* arow = &As[(wm * 32 + mt * 16 + lr) * ASTR];
      const int ko = hi ? 8 : 0;
#pragma unroll
      for (int j = 0; j < 8; ++j) {
        af[mt][j] = arow[ko + j];           // K = ko+j
        af[mt][j + 8] = arow[16 + ko + j];  // K = 16+ko+j
      }
    }
#pragma unroll
    for (int nt = 0; nt < 4; ++nt) {
      const _Float16* bcol =
          &Bt[(wn * 64 + nt * 16 + lr) * BSTRT + (hi ? 8 : 0)];
#pragma unroll
      for (int j = 0; j < 8; ++j) {
        bf[nt][j] = bcol[j];
        bf[nt][j + 8] = bcol[16 + j];
      }
    }
#pragma unroll
    for (int mt = 0; mt < 2; ++mt)
#pragma unroll
      for (int nt = 0; nt < 4; ++nt)
        acc[mt][nt] = __builtin_amdgcn_wmma_f32_16x16x32_f16(
            false, af[mt], false, bf[nt], (short)0, acc[mt][nt], false, false);
    __syncthreads();
  }

  // epilogue: scale, bias, relu, store (C layout: VGPR r -> M=r / r+8)
#pragma unroll
  for (int mt = 0; mt < 2; ++mt)
#pragma unroll
    for (int nt = 0; nt < 4; ++nt) {
      const long col = colBase + wn * 64 + nt * 16 + lr;
      const float bv = bias ? bias[col] : 0.f;
#pragma unroll
      for (int r = 0; r < 8; ++r) {
        const long row = rowBase + wm * 32 + mt * 16 + r + (hi ? 8 : 0);
        float o = acc[mt][nt][r] * scale + bv;
        if (RELU) o = fmaxf(o, 0.f);
        if (OUTF32)
          ((float*)Cp)[offC + row * ldc + col] = o;
        else
          ((_Float16*)Cp)[offC + row * ldc + col] = (_Float16)o;
      }
    }
}

static inline void launch_gemm(hipStream_t s, const void* A, const _Float16* B,
                               const float* bias, void* C, int M, int N, int K,
                               long lda, long ldb, long ldc, int batches,
                               int inner, long sA1, long sA2, long sB1,
                               long sB2, long sC1, long sC2, float scale,
                               bool af32, bool outf32, bool relu) {
  dim3 g(N / 128, M / 128, batches), b(256);
#define HG_CALL(a, o, r)                                                      \
  hgemm<a, o, r><<<g, b, 0, s>>>(A, B, bias, C, K, lda, ldb, ldc, inner, sA1, \
                                 sA2, sB1, sB2, sC1, sC2, scale)
  if (af32) {
    if (outf32) { if (relu) HG_CALL(true, true, true); else HG_CALL(true, true, false); }
    else        { if (relu) HG_CALL(true, false, true); else HG_CALL(true, false, false); }
  } else {
    if (outf32) { if (relu) HG_CALL(false, true, true); else HG_CALL(false, true, false); }
    else        { if (relu) HG_CALL(false, false, true); else HG_CALL(false, false, false); }
  }
#undef HG_CALL
}

// ===================== weight transpose+convert [K][N]f32 -> [N][K]f16 =====
__global__ __launch_bounds__(256) void wtrans_kernel(
    const float* __restrict__ in, _Float16* __restrict__ out, int K, int N) {
  __shared__ float tile[32][33];
  const long zoff = (long)blockIdx.z * K * N;
  const int bx = blockIdx.x * 32;  // n
  const int by = blockIdx.y * 32;  // k
  const int tx = threadIdx.x & 31, ty = threadIdx.x >> 5;  // 32x8
#pragma unroll
  for (int i = 0; i < 32; i += 8)
    tile[ty + i][tx] = in[zoff + (long)(by + ty + i) * N + bx + tx];
  __syncthreads();
#pragma unroll
  for (int i = 0; i < 32; i += 8)
    out[zoff + (long)(bx + ty + i) * K + by + tx] = (_Float16)tile[tx][ty + i];
}

// ===================== V transpose per graph: [n][D] -> [g][D][n] ==========
__global__ __launch_bounds__(256) void vtrans_kernel(
    const _Float16* __restrict__ v, _Float16* __restrict__ vT) {
  __shared__ _Float16 tile[32][33];
  const int g = blockIdx.z;
  const int bx = blockIdx.x * 32;  // d
  const int by = blockIdx.y * 32;  // n
  const int tx = threadIdx.x & 31, ty = threadIdx.x >> 5;
#pragma unroll
  for (int i = 0; i < 32; i += 8)
    tile[ty + i][tx] = v[((long)g * NPGC + by + ty + i) * DIM + bx + tx];
  __syncthreads();
#pragma unroll
  for (int i = 0; i < 32; i += 8)
    vT[(long)g * DIM * NPGC + (long)(bx + ty + i) * NPGC + by + tx] =
        tile[tx][ty + i];
}

// ===================== encoders (emit f16 activations) =====================
__global__ __launch_bounds__(256) void atom_encode(
    const float* __restrict__ emb, const int* __restrict__ x,
    _Float16* __restrict__ h) {
  const long idx = (long)blockIdx.x * 256 + threadIdx.x;  // N*D
  const long n = idx >> 9;
  const int d = idx & 511;
  float s = 0.f;
#pragma unroll
  for (int f = 0; f < AFN; ++f) {
    const int tok = x[n * AFN + f];
    s += emb[((long)f * AVN + tok) * DIM + d];
  }
  h[idx] = (_Float16)s;
}

__global__ __launch_bounds__(256) void bond_encode(
    const float* __restrict__ emb, const int* __restrict__ ea,
    _Float16* __restrict__ out) {
  const long idx = (long)blockIdx.x * 256 + threadIdx.x;  // E*D
  const long e = idx >> 9;
  const int d = idx & 511;
  float s = 0.f;
#pragma unroll
  for (int f = 0; f < BFN; ++f) {
    const int tok = ea[e * BFN + f];
    s += emb[((long)f * BVN + tok) * DIM + d];
  }
  out[idx] = (_Float16)s;
}

// ===================== graph attention (segment softmax) ===================
__device__ inline void atomicMaxF(float* addr, float v) {
  if (v >= 0.f)
    atomicMax((int*)addr, __float_as_int(v));
  else
    atomicMin((unsigned int*)addr, (unsigned int)__float_as_int(v));
}

__global__ __launch_bounds__(256) void gc_init(float* __restrict__ mbuf,
                                               float* __restrict__ denom) {
  const int idx = blockIdx.x * 256 + threadIdx.x;  // N*HG
  mbuf[idx] = -3.0e38f;
  denom[idx] = 0.f;
}

__global__ __launch_bounds__(256) void gc_logits(
    const _Float16* __restrict__ q, const _Float16* __restrict__ k,
    const _Float16* __restrict__ ee, const int* __restrict__ ei,
    float* __restrict__ logits, float* __restrict__ mbuf) {
  const long idx = (long)blockIdx.x * 256 + threadIdx.x;  // E*HG
  const long e = idx >> 3;
  const int hh = idx & 7;
  const int src = ei[e], dst = ei[NE + e];
  const _Float16* qp = q + (long)dst * DIM + hh * CGC;
  const _Float16* kp = k + (long)src * DIM + hh * CGC;
  const _Float16* ep = ee + e * DIM + hh * CGC;
  float s = 0.f;
#pragma unroll 8
  for (int c = 0; c < CGC; ++c)
    s += (float)qp[c] * ((float)kp[c] + (float)ep[c]);
  s *= 0.125f;  // 1/sqrt(64)
  logits[idx] = s;
  atomicMaxF(&mbuf[(long)dst * HGC + hh], s);
}

__global__ __launch_bounds__(256) void gc_expsum(const int* __restrict__ ei,
                                                 float* __restrict__ logits,
                                                 const float* __restrict__ mbuf,
                                                 float* __restrict__ denom) {
  const long idx = (long)blockIdx.x * 256 + threadIdx.x;  // E*HG
  const long e = idx >> 3;
  const int hh = idx & 7;
  const int dst = ei[NE + e];
  const float p = __expf(logits[idx] - mbuf[(long)dst * HGC + hh]);
  logits[idx] = p;
  atomicAdd(&denom[(long)dst * HGC + hh], p);
}

__global__ __launch_bounds__(256) void gc_scatter(
    const _Float16* __restrict__ vbuf, const _Float16* __restrict__ ee,
    const float* __restrict__ pbuf, const float* __restrict__ denom,
    const int* __restrict__ ei, float* __restrict__ hnew) {
  const long e = blockIdx.x;  // one block per edge
  const int t = threadIdx.x;
  const int src = ei[e], dst = ei[NE + e];
  __shared__ float alpha[HGC];
  if (t < HGC)
    alpha[t] = pbuf[e * HGC + t] / (denom[(long)dst * HGC + t] + 1e-16f);
  __syncthreads();
#pragma unroll
  for (int u = 0; u < 2; ++u) {
    const int d = t + u * 256;
    const float val = alpha[d >> 6] *
                      ((float)vbuf[(long)src * DIM + d] + (float)ee[e * DIM + d]);
    atomicAdd(&hnew[(long)dst * DIM + d], val);
  }
}

// ===================== layernorm (+residual, +relu), f16 in/out ============
__global__ __launch_bounds__(256) void ln_kernel(
    const _Float16* __restrict__ X, const _Float16* __restrict__ R,
    const float* __restrict__ g, const float* __restrict__ b,
    _Float16* __restrict__ out, int relu) {
  __shared__ float red[256];
  const long row = blockIdx.x;
  const int t = threadIdx.x;
  const _Float16* xr = X + row * DIM;
  float x0 = (float)xr[t], x1 = (float)xr[t + 256];
  if (R) {
    const _Float16* rr = R + row * DIM;
    x0 += (float)rr[t];
    x1 += (float)rr[t + 256];
  }
  red[t] = x0 + x1;
  __syncthreads();
  for (int ss = 128; ss > 0; ss >>= 1) {
    if (t < ss) red[t] += red[t + ss];
    __syncthreads();
  }
  const float mean = red[0] * (1.f / DIM);
  __syncthreads();
  const float d0 = x0 - mean, d1 = x1 - mean;
  red[t] = d0 * d0 + d1 * d1;
  __syncthreads();
  for (int ss = 128; ss > 0; ss >>= 1) {
    if (t < ss) red[t] += red[t + ss];
    __syncthreads();
  }
  const float rstd = rsqrtf(red[0] * (1.f / DIM) + 1e-5f);
  float o0 = d0 * rstd * g[t] + b[t];
  float o1 = d1 * rstd * g[t + 256] + b[t + 256];
  if (relu) {
    o0 = fmaxf(o0, 0.f);
    o1 = fmaxf(o1, 0.f);
  }
  _Float16* orow = out + row * DIM;
  orow[t] = (_Float16)o0;
  orow[t + 256] = (_Float16)o1;
}

// ===================== softmax over rows of 256 (f16) ======================
__global__ __launch_bounds__(256) void softmax_kernel(
    _Float16* __restrict__ attn) {
  __shared__ float red[256];
  _Float16* p = attn + (long)blockIdx.x * 256;
  const int t = threadIdx.x;
  const float v = (float)p[t];
  red[t] = v;
  __syncthreads();
  for (int ss = 128; ss > 0; ss >>= 1) {
    if (t < ss) red[t] = fmaxf(red[t], red[t + ss]);
    __syncthreads();
  }
  const float m = red[0];
  __syncthreads();
  const float e = __expf(v - m);
  red[t] = e;
  __syncthreads();
  for (int ss = 128; ss > 0; ss >>= 1) {
    if (t < ss) red[t] += red[t + ss];
    __syncthreads();
  }
  p[t] = (_Float16)(e / red[0]);
}

// ===================== mean pool (contiguous batches), f16 -> f32 ==========
__global__ __launch_bounds__(256) void pool_kernel(
    const _Float16* __restrict__ h, float* __restrict__ pooled) {
  const int idx = blockIdx.x * 256 + threadIdx.x;  // G*D
  const int g = idx >> 9, d = idx & 511;
  const _Float16* base = h + (long)g * NPGC * DIM + d;
  float s = 0.f;
  for (int i = 0; i < NPGC; ++i) s += (float)base[(long)i * DIM];
  pooled[idx] = s * (1.f / NPGC);
}

// =====================================================================
extern "C" void kernel_launch(void* const* d_in, const int* in_sizes, int n_in,
                              void* d_out, int out_size, void* d_ws,
                              size_t ws_size, hipStream_t stream) {
  (void)in_sizes; (void)n_in; (void)out_size; (void)ws_size;
  const float* atom_emb = (const float*)d_in[0];
  const float* bond_emb = (const float*)d_in[1];
  const float* gq_w = (const float*)d_in[2];
  const float* gk_w = (const float*)d_in[3];
  const float* gv_w = (const float*)d_in[4];
  const float* ge_w = (const float*)d_in[5];
  const float* gskip_w = (const float*)d_in[6];
  const float* mid_w = (const float*)d_in[7];
  const float* gq_b = (const float*)d_in[8];
  const float* gk_b = (const float*)d_in[9];
  const float* gv_b = (const float*)d_in[10];
  const float* gskip_b = (const float*)d_in[11];
  const float* mid_b = (const float*)d_in[12];
  const float* mid_g = (const float*)d_in[13];
  const float* mid_bt = (const float*)d_in[14];
  const float* tq_w = (const float*)d_in[15];
  const float* tk_w = (const float*)d_in[16];
  const float* tv_w = (const float*)d_in[17];
  const float* to_w = (const float*)d_in[18];
  const float* tq_b = (const float*)d_in[19];
  const float* tk_b = (const float*)d_in[20];
  const float* tv_b = (const float*)d_in[21];
  const float* to_b = (const float*)d_in[22];
  const float* ln1_g = (const float*)d_in[23];
  const float* ln1_b = (const float*)d_in[24];
  const float* f1_w = (const float*)d_in[25];
  const float* f1_b = (const float*)d_in[26];
  const float* f2_w = (const float*)d_in[27];
  const float* f2_b = (const float*)d_in[28];
  const float* ln2_g = (const float*)d_in[29];
  const float* ln2_b = (const float*)d_in[30];
  const float* out_w = (const float*)d_in[31];
  const float* out_b = (const float*)d_in[32];
  const int* xin = (const int*)d_in[33];
  const int* edge_attr = (const int*)d_in[34];
  const int* edge_index = (const int*)d_in[35];

  // ---- workspace carve-up (byte based, 256B aligned chunks) ----
  char* Wb = (char*)d_ws;
  auto takeB = [&](size_t bytes) {
    char* p = Wb;
    Wb += (bytes + 255) & ~(size_t)255;
    return p;
  };
  _Float16* h    = (_Float16*)takeB((size_t)NN * DIM * 2);
  _Float16* q    = (_Float16*)takeB((size_t)NN * DIM * 2);
  _Float16* k    = (_Float16*)takeB((size_t)NN * DIM * 2);
  _Float16* v    = (_Float16*)takeB((size_t)NN * DIM * 2);
  _Float16* t1   = (_Float16*)takeB((size_t)NN * DIM * 2);
  _Float16* vT   = (_Float16*)takeB((size_t)NN * DIM * 2);
  _Float16* attn = (_Float16*)takeB((size_t)NG * HTC * NPGC * NPGC * 2);
  _Float16* ff1  = (_Float16*)takeB((size_t)NN * FFD * 2);
  _Float16* eattr = (_Float16*)takeB((size_t)NE * DIM * 2);
  _Float16* ee    = (_Float16*)takeB((size_t)NE * DIM * 2);
  float* hnew   = (float*)takeB((size_t)NN * DIM * 4);
  float* logits = (float*)takeB((size_t)NE * HGC * 4);
  float* mbuf   = (float*)takeB((size_t)NN * HGC * 4);
  float* denom  = (float*)takeB((size_t)NN * HGC * 4);
  float* pooled = (float*)takeB((size_t)NG * DIM * 4);
  // transposed f16 weights
  const size_t WSQ = (size_t)NL * DIM * DIM;
  _Float16* gqT = (_Float16*)takeB(WSQ * 2);
  _Float16* gkT = (_Float16*)takeB(WSQ * 2);
  _Float16* gvT = (_Float16*)takeB(WSQ * 2);
  _Float16* geT = (_Float16*)takeB(WSQ * 2);
  _Float16* gsT = (_Float16*)takeB(WSQ * 2);
  _Float16* gmT = (_Float16*)takeB(WSQ * 2);
  _Float16* tqT = (_Float16*)takeB(WSQ * 2);
  _Float16* tkT = (_Float16*)takeB(WSQ * 2);
  _Float16* tvT = (_Float16*)takeB(WSQ * 2);
  _Float16* toT = (_Float16*)takeB(WSQ * 2);
  _Float16* f1T = (_Float16*)takeB((size_t)NL * DIM * FFD * 2);
  _Float16* f2T = (_Float16*)takeB((size_t)NL * DIM * FFD * 2);
  _Float16* outT = (_Float16*)takeB((size_t)DIM * OUTD * 2);

  // ---- one-time (per launch) weight convert+transpose ----
  auto wtr = [&](const float* w, _Float16* wt, int K_, int N_, int L_) {
    dim3 g(N_ / 32, K_ / 32, L_), b(256);
    wtrans_kernel<<<g, b, 0, stream>>>(w, wt, K_, N_);
  };
  wtr(gq_w, gqT, DIM, DIM, NL);
  wtr(gk_w, gkT, DIM, DIM, NL);
  wtr(gv_w, gvT, DIM, DIM, NL);
  wtr(ge_w, geT, DIM, DIM, NL);
  wtr(gskip_w, gsT, DIM, DIM, NL);
  wtr(mid_w, gmT, DIM, DIM, NL);
  wtr(tq_w, tqT, DIM, DIM, NL);
  wtr(tk_w, tkT, DIM, DIM, NL);
  wtr(tv_w, tvT, DIM, DIM, NL);
  wtr(to_w, toT, DIM, DIM, NL);
  wtr(f1_w, f1T, DIM, FFD, NL);
  wtr(f2_w, f2T, FFD, DIM, NL);
  wtr(out_w, outT, DIM, OUTD, 1);

  // encoders
  atom_encode<<<(long)NN * DIM / 256, 256, 0, stream>>>(atom_emb, xin, h);
  bond_encode<<<(long)NE * DIM / 256, 256, 0, stream>>>(bond_emb, edge_attr,
                                                        eattr);

  // ---------------- graph conv layers ----------------
  for (int i = 0; i < NL; ++i) {
    const long wof = (long)i * DIM * DIM, bof = (long)i * DIM;
    // ee = e_attr @ ge_w   (E x 512 x 512 — the dominant GEMM)
    launch_gemm(stream, eattr, geT + wof, nullptr, ee, NE, DIM, DIM, DIM, DIM,
                DIM, 1, 1, 0, 0, 0, 0, 0, 0, 1.f, false, false, false);
    launch_gemm(stream, h, gqT + wof, gq_b + bof, q, NN, DIM, DIM, DIM, DIM,
                DIM, 1, 1, 0, 0, 0, 0, 0, 0, 1.f, false, false, false);
    launch_gemm(stream, h, gkT + wof, gk_b + bof, k, NN, DIM, DIM, DIM, DIM,
                DIM, 1, 1, 0, 0, 0, 0, 0, 0, 1.f, false, false, false);
    launch_gemm(stream, h, gvT + wof, gv_b + bof, v, NN, DIM, DIM, DIM, DIM,
                DIM, 1, 1, 0, 0, 0, 0, 0, 0, 1.f, false, false, false);
    // hnew(f32) = root skip: h @ gskip_w + gskip_b ; scatter adds into it
    launch_gemm(stream, h, gsT + wof, gskip_b + bof, hnew, NN, DIM, DIM, DIM,
                DIM, DIM, 1, 1, 0, 0, 0, 0, 0, 0, 1.f, false, true, false);
    gc_init<<<NN * HGC / 256, 256, 0, stream>>>(mbuf, denom);
    gc_logits<<<(long)NE * HGC / 256, 256, 0, stream>>>(q, k, ee, edge_index,
                                                        logits, mbuf);
    gc_expsum<<<(long)NE * HGC / 256, 256, 0, stream>>>(edge_index, logits,
                                                        mbuf, denom);
    gc_scatter<<<NE, 256, 0, stream>>>(v, ee, logits, denom, edge_index, hnew);
    // mid linear (A=f32 hnew) + LN + ReLU -> h
    launch_gemm(stream, hnew, gmT + wof, mid_b + bof, t1, NN, DIM, DIM, DIM,
                DIM, DIM, 1, 1, 0, 0, 0, 0, 0, 0, 1.f, true, false, false);
    ln_kernel<<<NN, 256, 0, stream>>>(t1, nullptr, mid_g + bof, mid_bt + bof, h,
                                      1);
  }

  // ---------------- dense transformer layers (xg lives in h) ----------------
  for (int i = 0; i < NL; ++i) {
    const long wof = (long)i * DIM * DIM, bof = (long)i * DIM;
    launch_gemm(stream, h, tqT + wof, tq_b + bof, q, NN, DIM, DIM, DIM, DIM,
                DIM, 1, 1, 0, 0, 0, 0, 0, 0, 1.f, false, false, false);
    launch_gemm(stream, h, tkT + wof, tk_b + bof, k, NN, DIM, DIM, DIM, DIM,
                DIM, 1, 1, 0, 0, 0, 0, 0, 0, 1.f, false, false, false);
    launch_gemm(stream, h, tvT + wof, tv_b + bof, v, NN, DIM, DIM, DIM, DIM,
                DIM, 1, 1, 0, 0, 0, 0, 0, 0, 1.f, false, false, false);
    // vT[g][c][n] for the attn@V GEMM's unified B layout
    {
      dim3 gt(DIM / 32, NPGC / 32, NG), bt(256);
      vtrans_kernel<<<gt, bt, 0, stream>>>(v, vT);
    }
    // attn[g,h] = (Q K^T)/sqrt(CT): batched (G*HT); K is naturally [n][k]
    launch_gemm(stream, q, k, nullptr, attn, NPGC, NPGC, CTC, DIM, DIM, NPGC,
                NG * HTC, HTC, (long)NPGC * DIM, CTC, (long)NPGC * DIM, CTC,
                (long)HTC * NPGC * NPGC, (long)NPGC * NPGC, 0.08838834764f,
                false, false, false);
    softmax_kernel<<<NG * HTC * NPGC, 256, 0, stream>>>(attn);
    // o = attn @ V -> reuse q buffer (B = vT slice [c][n])
    launch_gemm(stream, attn, vT, nullptr, q, NPGC, CTC, NPGC, NPGC, NPGC, DIM,
                NG * HTC, HTC, (long)HTC * NPGC * NPGC, (long)NPGC * NPGC,
                (long)DIM * NPGC, (long)CTC * NPGC, (long)NPGC * DIM, CTC, 1.f,
                false, false, false);
    launch_gemm(stream, q, toT + wof, to_b + bof, t1, NN, DIM, DIM, DIM, DIM,
                DIM, 1, 1, 0, 0, 0, 0, 0, 0, 1.f, false, false, false);
    ln_kernel<<<NN, 256, 0, stream>>>(h, t1, ln1_g + bof, ln1_b + bof, h, 0);
    // FFN
    launch_gemm(stream, h, f1T + (long)i * DIM * FFD, f1_b + (long)i * FFD, ff1,
                NN, FFD, DIM, DIM, DIM, FFD, 1, 1, 0, 0, 0, 0, 0, 0, 1.f, false,
                false, true);
    launch_gemm(stream, ff1, f2T + (long)i * DIM * FFD, f2_b + bof, t1, NN, DIM,
                FFD, FFD, FFD, DIM, 1, 1, 0, 0, 0, 0, 0, 0, 1.f, false, false,
                false);
    ln_kernel<<<NN, 256, 0, stream>>>(h, t1, ln2_g + bof, ln2_b + bof, h, 0);
  }

  // readout + output projection (A = pooled f32, C = d_out f32)
  pool_kernel<<<NG * DIM / 256, 256, 0, stream>>>(h, pooled);
  launch_gemm(stream, pooled, outT, out_b, (float*)d_out, NG, OUTD, DIM, DIM,
              DIM, OUTD, 1, 1, 0, 0, 0, 0, 0, 0, 1.f, true, true, false);
}
